// SFNet_32547262169639
// MI455X (gfx1250) — matrix operand order repaired and checked by reading, not compiled, not run on hardware
//
#include <hip/hip_runtime.h>

typedef __attribute__((ext_vector_type(16))) __bf16 v16bf;
typedef __attribute__((ext_vector_type(8)))  float  v8f;
typedef __attribute__((ext_vector_type(4)))  unsigned int uv4;

#define HW 4096   // 64*64 pixels

// ---------------- per-pixel inverse L2 norm over channels ----------------
__global__ __launch_bounds__(256) void pixnorm_kernel(const float* __restrict__ f, int C,
                                                      float* __restrict__ invn)
{
    int i = blockIdx.x * 256 + threadIdx.x;           // pixel index, coalesced
    float ss = 0.f;
    for (int k = 0; k < C; ++k) {
        float v = f[(size_t)k * HW + i];
        ss = fmaf(v, v, ss);
    }
    invn[i] = rsqrtf(ss + 1e-6f);                     // x / sqrt(sum+1e-6)
}

// ---------- normalize + transpose (C, HW) f32 -> (HW, C) bf16 via LDS ----------
__global__ __launch_bounds__(256) void transpose_norm_kernel(const float* __restrict__ f,
                                                             const float* __restrict__ invn,
                                                             __bf16* __restrict__ T, int C)
{
    __shared__ float tile[32][33];
    int i0 = blockIdx.x * 32;                         // pixel tile
    int k0 = blockIdx.y * 32;                         // channel tile
    int tx = threadIdx.x & 31;
    int ty = threadIdx.x >> 5;                        // 8 rows per pass
#pragma unroll
    for (int r = 0; r < 32; r += 8)
        tile[ty + r][tx] = f[(size_t)(k0 + ty + r) * HW + i0 + tx];
    __syncthreads();
#pragma unroll
    for (int r = 0; r < 32; r += 8) {
        int i = i0 + ty + r;
        T[(size_t)i * C + k0 + tx] = (__bf16)(tile[tx][ty + r] * invn[i]);
    }
}

// ---------------- fragment loaders matching ISA 7.12.2 bf16 layouts ----------------
// A 16x32 bf16: lane<16 -> M=lane, K in {0..7, 16..23}; lane>=16 -> same M, K+8
__device__ inline void load_a_frag(const __bf16* base, v16bf& a)
{
    uv4* d = (uv4*)&a;
    d[0] = *(const uv4*)(base);                       // 8 bf16: K = klo..klo+7
    d[1] = *(const uv4*)(base + 16);                  // 8 bf16: K = klo+16..klo+23
}
// B 32x16 bf16: lane<16 -> N=lane, K in {0..7, 16..23}; lane>=16 -> same N, K+8
__device__ inline void load_b_frag(const __bf16* base, v16bf& b)
{
    uv4* d = (uv4*)&b;
    d[0] = *(const uv4*)(base);
    d[1] = *(const uv4*)(base + 16);
}

// ---------------- fused dual-GEMM: P = relu(t3T.s3T') * relu(t4T.s4T') ----------------
// t*T / s*T are (HW, C) bf16 row-major.  Writes P[j][i] and PT[i][j].
__global__ __launch_bounds__(256) void gemm_fused_kernel(const __bf16* __restrict__ t3T,
                                                         const __bf16* __restrict__ s3T,
                                                         const __bf16* __restrict__ t4T,
                                                         const __bf16* __restrict__ s4T,
                                                         float* __restrict__ P,
                                                         float* __restrict__ PT)
{
    const int lane = threadIdx.x & 31;
    const int wave = threadIdx.x >> 5;
    const int j0 = blockIdx.y * 128 + (wave >> 2) * 64;   // tgt pixel (M) base for this wave
    const int i0 = blockIdx.x * 128 + (wave & 3) * 32;    // src pixel (N) base for this wave
    const int mr = lane & 15;
    const int hi = lane >> 4;
    const int kA = hi * 8;                                // A half-wave K offset
    const int kB = hi * 8;                                // B half-wave K offset

    v8f acc3[4][2], acc4[4][2];
    v8f z = {};
#pragma unroll
    for (int mi = 0; mi < 4; ++mi)
#pragma unroll
        for (int ni = 0; ni < 2; ++ni) { acc3[mi][ni] = z; acc4[mi][ni] = z; }

    // ---- corr3: K = 1024 ----
    for (int k = 0; k < 1024; k += 32) {
        v16bf a[4], b[2];
#pragma unroll
        for (int mi = 0; mi < 4; ++mi)
            load_a_frag(t3T + (size_t)(j0 + mi * 16 + mr) * 1024 + k + kA, a[mi]);
#pragma unroll
        for (int ni = 0; ni < 2; ++ni)
            load_b_frag(s3T + (size_t)(i0 + ni * 16 + mr) * 1024 + k + kB, b[ni]);
        if (k + 32 < 1024) {
            __builtin_prefetch(t3T + (size_t)(j0 + mr) * 1024 + k + 32, 0, 3);
            __builtin_prefetch(s3T + (size_t)(i0 + mr) * 1024 + k + 32, 0, 3);
        }
#pragma unroll
        for (int mi = 0; mi < 4; ++mi)
#pragma unroll
            for (int ni = 0; ni < 2; ++ni)
                acc3[mi][ni] = __builtin_amdgcn_wmma_f32_16x16x32_bf16(
                    false, a[mi], false, b[ni], (short)0, acc3[mi][ni], false, false);
    }

    // ---- corr4: K = 2048 ----
    for (int k = 0; k < 2048; k += 32) {
        v16bf a[4], b[2];
#pragma unroll
        for (int mi = 0; mi < 4; ++mi)
            load_a_frag(t4T + (size_t)(j0 + mi * 16 + mr) * 2048 + k + kA, a[mi]);
#pragma unroll
        for (int ni = 0; ni < 2; ++ni)
            load_b_frag(s4T + (size_t)(i0 + ni * 16 + mr) * 2048 + k + kB, b[ni]);
        if (k + 32 < 2048) {
            __builtin_prefetch(t4T + (size_t)(j0 + mr) * 2048 + k + 32, 0, 3);
            __builtin_prefetch(s4T + (size_t)(i0 + mr) * 2048 + k + 32, 0, 3);
        }
#pragma unroll
        for (int mi = 0; mi < 4; ++mi)
#pragma unroll
            for (int ni = 0; ni < 2; ++ni)
                acc4[mi][ni] = __builtin_amdgcn_wmma_f32_16x16x32_bf16(
                    false, a[mi], false, b[ni], (short)0, acc4[mi][ni], false, false);
    }

    // ---- epilogue: relu * relu, dual store (C/D layout: m = e + hi*8, n = mr) ----
#pragma unroll
    for (int mi = 0; mi < 4; ++mi)
#pragma unroll
        for (int ni = 0; ni < 2; ++ni)
#pragma unroll
            for (int e = 0; e < 8; ++e) {
                float c3 = acc3[mi][ni][e]; c3 = c3 > 0.f ? c3 : 0.f;
                float c4 = acc4[mi][ni][e]; c4 = c4 > 0.f ? c4 : 0.f;
                float d = c3 * c4;
                int j = j0 + mi * 16 + hi * 8 + e;
                int i = i0 + ni * 16 + mr;
                P [(size_t)j * HW + i] = d;
                PT[(size_t)i * HW + j] = d;
            }
}

// ---------------- wave32 reductions ----------------
__device__ inline float wave_sum(float v)
{
#pragma unroll
    for (int o = 16; o; o >>= 1) v += __shfl_xor(v, o, 32);
    return v;
}
__device__ inline float wave_max(float v)
{
#pragma unroll
    for (int o = 16; o; o >>= 1) v = fmaxf(v, __shfl_xor(v, o, 32));
    return v;
}
__device__ inline void wave_argmax(float& v, int& idx)
{
#pragma unroll
    for (int o = 16; o; o >>= 1) {
        float ov = __shfl_xor(v, o, 32);
        int   oi = __shfl_xor(idx, o, 32);
        if (ov > v || (ov == v && oi < idx)) { v = ov; idx = oi; }   // first-max tie break
    }
}

// --------- L2-normalize (over row) + kernel-soft-argmax; one wave per row ---------
__global__ __launch_bounds__(256) void softargmax_kernel(const float* __restrict__ M,
                                                         float* __restrict__ grid_out)
{
    const int row  = blockIdx.x * 8 + (threadIdx.x >> 5);
    const int lane = threadIdx.x & 31;
    const float* r = M + (size_t)row * HW;

    // pass 1: sum of squares (for the axis-1 l2_normalize) + argmax
    float ss = 0.f, best = -1e30f;
    int bidx = 0;
    for (int j = lane; j < HW; j += 32) {
        float v = r[j];
        ss = fmaf(v, v, ss);
        if (v > best) { best = v; bidx = j; }
    }
    ss = wave_sum(ss);
    wave_argmax(best, bidx);
    float inv = rsqrtf(ss + 1e-6f);
    float ix = (float)(bidx & 63), iy = (float)(bidx >> 6);

    // pass 2: max of gauss * corr (softmax stabilizer)
    float m = -1e30f;
    for (int j = lane; j < HW; j += 32) {
        float jx = (float)(j & 63), jy = (float)(j >> 6);
        float dx = jx - ix, dy = jy - iy;
        float g = __expf(-(dx * dx + dy * dy) * 0.02f);   // 1/(2*sigma^2), sigma=5
        m = fmaxf(m, g * r[j] * inv);
    }
    m = wave_max(m);

    // pass 3: softmax(beta=50) expectations of normalized coords
    float s = 0.f, sx = 0.f, sy = 0.f;
    for (int j = lane; j < HW; j += 32) {
        float jx = (float)(j & 63), jy = (float)(j >> 6);
        float dx = jx - ix, dy = jy - iy;
        float g = __expf(-(dx * dx + dy * dy) * 0.02f);
        float v = g * r[j] * inv;
        float e = __expf(50.f * (v - m));
        s += e;
        sx = fmaf(e, -1.f + jx * (2.f / 63.f), sx);
        sy = fmaf(e, -1.f + jy * (2.f / 63.f), sy);
    }
    s = wave_sum(s); sx = wave_sum(sx); sy = wave_sum(sy);
    if (lane == 0) {
        grid_out[(size_t)row * 2 + 0] = sx / s;
        grid_out[(size_t)row * 2 + 1] = sy / s;
    }
}

// flows are identically zero in the reference (grid_x - grid_x)
__global__ __launch_bounds__(256) void zero_flow_kernel(float* __restrict__ out)
{
    int t = blockIdx.x * 256 + threadIdx.x;           // 16384 threads
    out[16384 + t] = 0.f;                             // flow_s2t
    out[49152 + t] = 0.f;                             // flow_t2s
}

extern "C" void kernel_launch(void* const* d_in, const int* in_sizes, int n_in,
                              void* d_out, int out_size, void* d_ws, size_t ws_size,
                              hipStream_t stream)
{
    (void)in_sizes; (void)n_in; (void)out_size; (void)ws_size;
    const float* src3 = (const float*)d_in[0];
    const float* tgt3 = (const float*)d_in[1];
    const float* src4 = (const float*)d_in[2];
    const float* tgt4 = (const float*)d_in[3];
    float* out = (float*)d_out;

    char* ws = (char*)d_ws;
    size_t off = 0;
    auto carve = [&](size_t bytes) -> char* {
        char* p = ws + off;
        off = (off + bytes + 255) & ~(size_t)255;
        return p;
    };
    __bf16* s3T = (__bf16*)carve((size_t)HW * 1024 * 2);
    __bf16* t3T = (__bf16*)carve((size_t)HW * 1024 * 2);
    __bf16* s4T = (__bf16*)carve((size_t)HW * 2048 * 2);
    __bf16* t4T = (__bf16*)carve((size_t)HW * 2048 * 2);
    float*  P   = (float*)carve((size_t)HW * HW * 4);
    float*  PT  = (float*)carve((size_t)HW * HW * 4);
    float*  inv_s3 = (float*)carve(HW * 4);
    float*  inv_t3 = (float*)carve(HW * 4);
    float*  inv_s4 = (float*)carve(HW * 4);
    float*  inv_t4 = (float*)carve(HW * 4);

    zero_flow_kernel<<<64, 256, 0, stream>>>(out);

    for (int b = 0; b < 2; ++b) {
        const float* s3 = src3 + (size_t)b * 1024 * HW;
        const float* t3 = tgt3 + (size_t)b * 1024 * HW;
        const float* s4 = src4 + (size_t)b * 2048 * HW;
        const float* t4 = tgt4 + (size_t)b * 2048 * HW;

        pixnorm_kernel<<<16, 256, 0, stream>>>(s3, 1024, inv_s3);
        pixnorm_kernel<<<16, 256, 0, stream>>>(t3, 1024, inv_t3);
        pixnorm_kernel<<<16, 256, 0, stream>>>(s4, 2048, inv_s4);
        pixnorm_kernel<<<16, 256, 0, stream>>>(t4, 2048, inv_t4);

        transpose_norm_kernel<<<dim3(128, 32), 256, 0, stream>>>(s3, inv_s3, s3T, 1024);
        transpose_norm_kernel<<<dim3(128, 32), 256, 0, stream>>>(t3, inv_t3, t3T, 1024);
        transpose_norm_kernel<<<dim3(128, 64), 256, 0, stream>>>(s4, inv_s4, s4T, 2048);
        transpose_norm_kernel<<<dim3(128, 64), 256, 0, stream>>>(t4, inv_t4, t4T, 2048);

        gemm_fused_kernel<<<dim3(32, 32), 256, 0, stream>>>(t3T, s3T, t4T, s4T, P, PT);

        // s2t: normalize/reduce over j == rows of PT; grid_s2t at out[0 + b*8192]
        softargmax_kernel<<<512, 256, 0, stream>>>(PT, out + (size_t)b * 8192);
        // t2s: corr_t2s[j,i] = P[i,j] -> rows of P; grid_t2s at out[32768 + b*8192]
        softargmax_kernel<<<512, 256, 0, stream>>>(P, out + 32768 + (size_t)b * 8192);
    }
}
